// RNN_9363028705408
// MI455X (gfx1250) — matrix operand order repaired and checked by reading, not compiled, not run on hardware
//
#include <hip/hip_runtime.h>
#include <hip/hip_bf16.h>

// RNN forward for MI455X (gfx1250), wave32 + WMMA bf16 path.
// BATCH=64, SEQ=2048, IN_DIM=HID=512.

#define BATCH  64
#define SEQ    2048
#define IN_DIM 512
#define HID    512

#define NWG_REC 16          // persistent recurrence workgroups (16 WGs x 8 waves = 128 tiles)
#define LDS_STRIDE 520      // 512 + 8 bf16 pad -> conflict-free A-fragment reads

typedef __bf16 bf16_t;
typedef __attribute__((ext_vector_type(16))) __bf16 v16bf;
typedef __attribute__((ext_vector_type(8)))  __bf16 v8bf;
typedef __attribute__((ext_vector_type(8)))  float  v8f;
typedef __attribute__((ext_vector_type(4)))  int    v4i;

#if __has_builtin(__builtin_amdgcn_global_load_async_to_lds_b128) && \
    __has_builtin(__builtin_amdgcn_s_wait_asynccnt)
#define HAVE_ASYNC_LDS 1
// builtin signature (from hipcc diagnostic): param0 = v4i in AS(1) ("__device__"),
// param1 = v4i in AS(3), then imm offset, imm cpol.
typedef __attribute__((address_space(1))) v4i gv4i_t;
typedef __attribute__((address_space(3))) v4i lv4i_t;
#else
#define HAVE_ASYNC_LDS 0
#endif

// workspace layout (bytes)
#define WS_WIH   0            // 512*512 bf16 = 512 KB
#define WS_WHH   524288       // 512*512 bf16 = 512 KB
#define WS_BIAS  1048576      // 512 f32
#define WS_HBUF  1052672      // 2 * 64*512 bf16 = 128 KB (double-buffered h)
#define WS_SYNC  1183744      // 2 ints: arrival counter, generation

// ---------------------------------------------------------------------------
// Setup: fp32 -> bf16 weight conversion, bias fusion, h0 -> hbuf[0], barrier reset.
// Runs every launch so graph replays are deterministic.
// ---------------------------------------------------------------------------
__global__ __launch_bounds__(256) void rnn_setup_kernel(
    const float* __restrict__ wih, const float* __restrict__ whh,
    const float* __restrict__ bih, const float* __restrict__ bhh,
    const float* __restrict__ h0,
    bf16_t* __restrict__ wih_b, bf16_t* __restrict__ whh_b,
    float* __restrict__ bias, bf16_t* __restrict__ hbuf,
    int* __restrict__ sync) {
  int g = blockIdx.x * blockDim.x + threadIdx.x;
  if (g < HID * IN_DIM) {
    wih_b[g] = (bf16_t)wih[g];
    whh_b[g] = (bf16_t)whh[g];
  }
  if (g < HID)         bias[g] = bih[g] + bhh[g];
  if (g < BATCH * HID) hbuf[g] = (bf16_t)h0[g];
  if (g < 2)           sync[g] = 0;
}

// ---------------------------------------------------------------------------
// Phase 1: gates = X (131072x512) * W_ih^T (512x512) + bias, written straight
// into d_out at (b,s,h). Memory-bound (~512 MB @ 23.3 TB/s ~= 22us floor), so
// bf16 WMMA with f32 accumulate is free accuracy-wise and halves W traffic.
// One 16x16 tile per wave; 16 K-steps of v_wmma_f32_16x16x32_bf16.
// ---------------------------------------------------------------------------
__global__ __launch_bounds__(256) void rnn_gate_gemm_kernel(
    const float*  __restrict__ x,       // inputs, rows flat (b*SEQ+s), 512 f32 each
    const bf16_t* __restrict__ wih_b,   // (HID, IN_DIM) row-major bf16
    const float*  __restrict__ bias,
    float*        __restrict__ out) {   // (b,s,h) == rows (b*SEQ+s) x 512
  const int lane = threadIdx.x & 31;
  const int wave = threadIdx.x >> 5;
  const int gid  = blockIdx.x * 8 + wave;   // tile id, 262144 total
  const int mt   = gid >> 5;                // 0..8191 (M tiles)
  const int nt   = gid & 31;                // 0..31   (N tiles)
  const int hi   = lane >> 4;               // lane group (K offset select)
  const int l16  = lane & 15;

  const int arow = mt * 16 + l16;           // A row = flat (b*SEQ+s)
  const int ncol = nt * 16 + l16;           // B/C column

  // C init: bias broadcast down the column
  const float bv = bias[ncol];
  v8f c;
#pragma unroll
  for (int r = 0; r < 8; ++r) c[r] = bv;

  const float*  ap = x     + (size_t)arow * IN_DIM;
  const bf16_t* bp = wih_b + (size_t)ncol * IN_DIM;  // B = W^T -> row n of W is a B column

#pragma unroll
  for (int k0 = 0; k0 < IN_DIM; k0 += 32) {
    // A fragment: lane<16 -> K {k0..+7, k0+16..+23}; lane>=16 -> +8 (ISA 16-bit A layout)
    v8f alo = *(const v8f*)(ap + k0 + hi * 8);
    v8f ahi = *(const v8f*)(ap + k0 + 16 + hi * 8);
    v16bf a;
#pragma unroll
    for (int e = 0; e < 8; ++e) { a[e] = (bf16_t)alo[e]; a[e + 8] = (bf16_t)ahi[e]; }

    // B fragment: W row ncol at K {k0+hi*8..+7, k0+16+hi*8..+7} (matches A layout)
    v8bf blo = *(const v8bf*)(bp + k0 + hi * 8);
    v8bf bhi = *(const v8bf*)(bp + k0 + 16 + hi * 8);
    v16bf b;
#pragma unroll
    for (int e = 0; e < 8; ++e) { b[e] = blo[e]; b[e + 8] = bhi[e]; }

    c = __builtin_amdgcn_wmma_f32_16x16x32_bf16(false, a, false, b, (short)0, c,
                                                false, false);
  }

  // C/D layout: VGPR r -> row mt*16 + r + 8*hi, col ncol
#pragma unroll
  for (int r = 0; r < 8; ++r) {
    int m = mt * 16 + r + hi * 8;
    out[(size_t)m * HID + ncol] = c[r];
  }
}

// ---------------------------------------------------------------------------
// Phase 2: persistent recurrence. 16 WGs x 8 waves = 128 tiles = full 64x512 h.
// W_hh B-fragments are hoisted into 128 VGPRs/lane ONCE (weight-stationary for
// all 2048 steps). Per step: async-stage h rows to LDS (ASYNCcnt path when the
// toolchain has the builtins), prefetch next step's gate tile (HBM latency is
// on the critical path), C = gate tile, 16x v_wmma_f32_16x16x32_bf16, relu,
// store, then a 16-arrival global barrier (monotone generation counter).
// ---------------------------------------------------------------------------
__global__ __launch_bounds__(256) void rnn_recurrence_kernel(
    const bf16_t* __restrict__ whh_b,  // (HID, HID) row-major bf16
    bf16_t*       __restrict__ hbuf,   // [2][BATCH*HID] bf16 double buffer
    float*        __restrict__ out,    // gates in / outputs out, (b,s,h)
    float*        __restrict__ hn,     // d_out tail: (BATCH, HID) final state
    int*          __restrict__ sync) { // [0]=arrival count, [1]=generation
  __shared__ bf16_t hs[16 * LDS_STRIDE];

  const int lane = threadIdx.x & 31;
  const int wave = threadIdx.x >> 5;
  const int mt   = blockIdx.x & 3;          // batch tile (rows m0..m0+15)
  const int ng   = blockIdx.x >> 2;         // n-tile group
  const int nt   = ng * 8 + wave;           // 0..31
  const int m0   = mt * 16;
  const int hi   = lane >> 4;
  const int l16  = lane & 15;
  const int ncol = nt * 16 + l16;

  // ---- weight-stationary: preload all 16 W_hh B-fragments into VGPRs ----
  const bf16_t* bp = whh_b + (size_t)ncol * HID;  // B = W_hh^T -> contiguous row reads
  v16bf bfr[16];
#pragma unroll
  for (int kk = 0; kk < 16; ++kk) {
    const bf16_t* p = bp + kk * 32 + hi * 8;
    v8bf blo = *(const v8bf*)(p);
    v8bf bhi = *(const v8bf*)(p + 16);
#pragma unroll
    for (int e = 0; e < 8; ++e) { bfr[kk][e] = blo[e]; bfr[kk][e + 8] = bhi[e]; }
  }

  for (int s = 0; s < SEQ; ++s) {
    const bf16_t* hread  = hbuf + (size_t)(s & 1) * (BATCH * HID);
    bf16_t*       hwrite = hbuf + (size_t)((s + 1) & 1) * (BATCH * HID);

    // stage h rows m0..m0+15 (16 KB) into LDS, padded stride
#if HAVE_ASYNC_LDS
#pragma unroll
    for (int j = 0; j < 4; ++j) {
      int e8  = threadIdx.x + 256 * j;   // chunk-of-8 bf16 (16 B), 1024 chunks
      int idx = e8 * 8;
      int row = idx >> 9;
      int col = idx & 511;
      gv4i_t* g = (gv4i_t*)(uintptr_t)(const void*)
          (hread + (size_t)(m0 + row) * HID + col);
      lv4i_t* l = (lv4i_t*)(uintptr_t)(uint32_t)(uintptr_t)
          (&hs[row * LDS_STRIDE + col]);           // generic low 32b == LDS offset
      __builtin_amdgcn_global_load_async_to_lds_b128(g, l, 0, 0);
    }
    __builtin_amdgcn_s_wait_asynccnt(0);
#else
#pragma unroll
    for (int j = 0; j < 4; ++j) {
      int e8  = threadIdx.x + 256 * j;
      int idx = e8 * 8;
      int row = idx >> 9;
      int col = idx & 511;
      *(v8bf*)(&hs[row * LDS_STRIDE + col]) =
          *(const v8bf*)(hread + (size_t)(m0 + row) * HID + col);
    }
#endif
    __syncthreads();

    // prefetch next step's gate tile into cache (hides HBM latency of C init)
    if (s + 1 < SEQ) {
#pragma unroll
      for (int r = 0; r < 8; ++r) {
        int b = m0 + r + hi * 8;
        __builtin_prefetch(&out[((size_t)b * SEQ + (s + 1)) * HID + ncol], 0, 3);
      }
    }

    // C init = gate tile, read in place from d_out
    v8f c;
#pragma unroll
    for (int r = 0; r < 8; ++r) {
      int b = m0 + r + hi * 8;
      c[r] = out[((size_t)b * SEQ + s) * HID + ncol];
    }

#pragma unroll
    for (int kk = 0; kk < 16; ++kk) {
      const bf16_t* lp = &hs[l16 * LDS_STRIDE + kk * 32 + hi * 8];
      v8bf alo = *(const v8bf*)(lp);
      v8bf ahi = *(const v8bf*)(lp + 16);
      v16bf a;
#pragma unroll
      for (int e = 0; e < 8; ++e) { a[e] = alo[e]; a[e + 8] = ahi[e]; }
      c = __builtin_amdgcn_wmma_f32_16x16x32_bf16(false, a, false, bfr[kk],
                                                  (short)0, c, false, false);
    }

    // relu + stores: outputs (f32, in place), next-h (bf16), final state
#pragma unroll
    for (int r = 0; r < 8; ++r) {
      int b = m0 + r + hi * 8;
      float v = c[r] < 0.0f ? 0.0f : c[r];
      out[((size_t)b * SEQ + s) * HID + ncol] = v;
      hwrite[(size_t)b * HID + ncol] = (bf16_t)v;
      if (s == SEQ - 1) hn[(size_t)b * HID + ncol] = v;
    }

    // ---- device-wide step barrier (16 arrivals, monotone generation) ----
    __threadfence();
    __syncthreads();
    if (threadIdx.x == 0) {
      int old = __hip_atomic_fetch_add(&sync[0], 1, __ATOMIC_ACQ_REL,
                                       __HIP_MEMORY_SCOPE_AGENT);
      if (old == NWG_REC * (s + 1) - 1) {
        __hip_atomic_store(&sync[1], s + 1, __ATOMIC_RELEASE,
                           __HIP_MEMORY_SCOPE_AGENT);
      } else {
        while (__hip_atomic_load(&sync[1], __ATOMIC_ACQUIRE,
                                 __HIP_MEMORY_SCOPE_AGENT) < s + 1) {
          __builtin_amdgcn_s_sleep(1);
        }
      }
    }
    __syncthreads();
  }
}

// ---------------------------------------------------------------------------
extern "C" void kernel_launch(void* const* d_in, const int* in_sizes, int n_in,
                              void* d_out, int out_size, void* d_ws, size_t ws_size,
                              hipStream_t stream) {
  const float* x   = (const float*)d_in[0];  // (64, 2048, 512)
  const float* h0  = (const float*)d_in[1];  // (1, 64, 512)
  const float* wih = (const float*)d_in[2];  // (512, 512)
  const float* whh = (const float*)d_in[3];  // (512, 512)
  const float* bih = (const float*)d_in[4];  // (512,)
  const float* bhh = (const float*)d_in[5];  // (512,)

  float* out = (float*)d_out;                          // outputs (64,2048,512)
  float* hn  = out + (size_t)BATCH * SEQ * HID;        // then h_n (1,64,512)

  char* ws = (char*)d_ws;
  bf16_t* wih_b = (bf16_t*)(ws + WS_WIH);
  bf16_t* whh_b = (bf16_t*)(ws + WS_WHH);
  float*  bias  = (float*)(ws + WS_BIAS);
  bf16_t* hbuf  = (bf16_t*)(ws + WS_HBUF);
  int*    syncp = (int*)(ws + WS_SYNC);

  // 1) convert weights/h0, fuse biases, reset barrier (every call)
  rnn_setup_kernel<<<(HID * IN_DIM + 255) / 256, 256, 0, stream>>>(
      wih, whh, bih, bhh, h0, wih_b, whh_b, bias, hbuf, syncp);

  // 2) gate GEMM: 262144 tiles, 8 waves/block -> 32768 blocks
  rnn_gate_gemm_kernel<<<32768, 256, 0, stream>>>(x, wih_b, bias, out);

  // 3) persistent recurrence: 16 resident workgroups, 2048 globally-synced steps
  rnn_recurrence_kernel<<<NWG_REC, 256, 0, stream>>>(whh_b, hbuf, out, hn, syncp);
}
